// CapsuleNet_79748952752185
// MI455X (gfx1250) — matrix operand order repaired and checked by reading, not compiled, not run on hardware
//
#include <hip/hip_runtime.h>

typedef __attribute__((ext_vector_type(16))) _Float16 v16h;
typedef __attribute__((ext_vector_type(8)))  float    v8f;

#define NB     128
#define ICAPS  2048
#define OCAPS  32
#define ODIM   16
#define KK_TOT 16384   // ICAPS * 8
#define KSTEPS 512     // KK_TOT / 32

// ---------------------------------------------------------------------------
// Kernel 1: PrimaryCapsule conv (8x8 per position) + bias + squash(dim=8).
// Output transposed: xsT[kk][b], kk = i*8 + k  (GEMM B-operand friendly).
// ---------------------------------------------------------------------------
__global__ void k_primary(const float* __restrict__ hidden,
                          const float* __restrict__ cw,
                          const float* __restrict__ cb,
                          float* __restrict__ xsT) {
  int idx = blockIdx.x * blockDim.x + threadIdx.x;   // B*ICAPS = 262144
  if (idx >= NB * ICAPS) return;
  int i = idx & (ICAPS - 1);
  int b = idx >> 11;
  float in[8];
#pragma unroll
  for (int k = 0; k < 8; ++k) in[k] = hidden[(size_t)b * 16384 + k * 2048 + i];
  float out[8];
  float nrm2 = 0.f;
#pragma unroll
  for (int c = 0; c < 8; ++c) {
    float acc = cb[c];
#pragma unroll
    for (int k = 0; k < 8; ++k) acc += cw[c * 8 + k] * in[k];
    out[c] = acc;
    nrm2 += acc * acc;
  }
  float n = sqrtf(nrm2);
  float sc = nrm2 / (1.f + nrm2) / (n + 1e-8f);
#pragma unroll
  for (int c = 0; c < 8; ++c) xsT[(size_t)(i * 8 + c) * NB + b] = out[c] * sc;
}

// ---------------------------------------------------------------------------
// Kernel 2: pack caps_w (f32 [O,I,16,8]) into WMMA A-operand f16 layout:
// Wh[o][kstep][lane][e], lane: d = lane&15, hi = lane>>4,
// per-lane element e -> K = e + 8*hi + (e>=8 ? 8 : 0), kk = kstep*32 + K.
// ---------------------------------------------------------------------------
__global__ void k_pack_w(const float* __restrict__ W, _Float16* __restrict__ Wh) {
  int idx = blockIdx.x * blockDim.x + threadIdx.x;   // 32*512*32 = 524288
  if (idx >= OCAPS * KSTEPS * 32) return;
  int lane  = idx & 31;
  int kstep = (idx >> 5) & (KSTEPS - 1);
  int o     = idx >> 14;
  int d  = lane & 15;
  int hi = lane >> 4;
  v16h vals;
#pragma unroll
  for (int e = 0; e < 16; ++e) {
    int K  = e + 8 * hi + ((e >= 8) ? 8 : 0);
    int kk = kstep * 32 + K;
    int i  = kk >> 3;
    int k  = kk & 7;
    vals[e] = (_Float16)W[((size_t)(o * ICAPS + i) * ODIM + d) * 8 + k];
  }
  *(v16h*)(Wh + (size_t)idx * 16) = vals;
}

// ---------------------------------------------------------------------------
// B-operand former: lane&15 = b column, 16 halfs = K rows in WMMA table order.
// ---------------------------------------------------------------------------
template <int WEIGHTED>
__device__ __forceinline__ v16h make_b(const float* __restrict__ xrow,
                                       const float* __restrict__ crow) {
  v16h bv;
  if (WEIGHTED) {
#pragma unroll
    for (int e = 0; e < 16; ++e) bv[e] = (_Float16)(xrow[(e + ((e >= 8) ? 8 : 0)) * NB] * crow[((e >= 8) ? 2 : 0) * NB]);
  } else {
#pragma unroll
    for (int e = 0; e < 16; ++e) bv[e] = (_Float16)xrow[(e + ((e >= 8) ? 8 : 0)) * NB];
  }
  return bv;
}

// ---------------------------------------------------------------------------
// Kernel 3: WMMA GEMM per output capsule o:
//   s[b,o,d] = sum_kk Wflat[o,d,kk] * (c[b,o,i(kk)] * x[b,kk])
// One workgroup = (o, btile): 8 waves split the 512 K-steps, LDS reduce.
// Two independent accumulators (even/odd K-step) so back-to-back WMMAs have
// no D->C dependency chain and address/convert VALU co-executes under WMMA.
// ---------------------------------------------------------------------------
template <int WEIGHTED>
__global__ void k_gemm_s(const _Float16* __restrict__ Wh,
                         const float* __restrict__ xsT,
                         const float* __restrict__ cT,
                         float* __restrict__ s) {
  __shared__ float red[8 * 32 * 8];
  int o     = blockIdx.x >> 3;
  int btile = blockIdx.x & 7;
  int wave  = threadIdx.x >> 5;
  int lane  = threadIdx.x & 31;
  int bcol  = btile * 16;
  int k0    = wave * 64;                       // this wave's first K-step
  // Per-K-step strides: A = 32 lanes * 16 halfs = 512 halfs;
  // xsT = 32 rows * 128 = 4096 floats; cT = 4 capsules * 128 = 512 floats.
  const _Float16* aptr = Wh + ((size_t)o * KSTEPS * 32 + (size_t)k0 * 32 + lane) * 16;
  const float*    xptr = xsT + ((size_t)k0 * 32 + 8 * (lane >> 4)) * NB + bcol + (lane & 15);
  const float*    cptr = cT + ((size_t)o * ICAPS + k0 * 4 + (lane >> 4)) * NB + bcol + (lane & 15);
  v8f acc0 = {};
  v8f acc1 = {};
  for (int t = 0; t < 32; ++t) {
    v16h a0 = *(const v16h*)aptr;
    v16h a1 = *(const v16h*)(aptr + 512);
    v16h b0 = make_b<WEIGHTED>(xptr, cptr);
    v16h b1 = make_b<WEIGHTED>(xptr + 4096, cptr + 512);
    acc0 = __builtin_amdgcn_wmma_f32_16x16x32_f16(false, a0, false, b0,
                                                  (short)0, acc0, false, false);
    acc1 = __builtin_amdgcn_wmma_f32_16x16x32_f16(false, a1, false, b1,
                                                  (short)0, acc1, false, false);
    aptr += 1024;
    xptr += 8192;
    cptr += 1024;
  }
#pragma unroll
  for (int r = 0; r < 8; ++r) red[(wave * 32 + lane) * 8 + r] = acc0[r] + acc1[r];
  __syncthreads();
  if (wave == 0) {
    int n  = lane & 15;
    int hi = lane >> 4;
    int b  = bcol + n;
#pragma unroll
    for (int r = 0; r < 8; ++r) {
      float v = 0.f;
#pragma unroll
      for (int w2 = 0; w2 < 8; ++w2) v += red[(w2 * 32 + lane) * 8 + r];
      int d = r + 8 * hi;
      s[((size_t)b * OCAPS + o) * ODIM + d] = v;
    }
  }
}

// ---------------------------------------------------------------------------
// Kernel 4: squash over d=16; final_len mode writes ||squash(s)|| (the output).
// scale folds the uniform c = 1/32 of routing iteration 0.
// ---------------------------------------------------------------------------
__global__ void k_squash(const float* __restrict__ s, float* __restrict__ vout,
                         float scale, int final_len) {
  int idx = blockIdx.x * blockDim.x + threadIdx.x;   // B*OCAPS = 4096
  if (idx >= NB * OCAPS) return;
  const float* sp = s + (size_t)idx * ODIM;
  float tmp[ODIM];
  float nrm2 = 0.f;
#pragma unroll
  for (int d = 0; d < ODIM; ++d) {
    tmp[d] = sp[d] * scale;
    nrm2 += tmp[d] * tmp[d];
  }
  float n  = sqrtf(nrm2);
  float sc = nrm2 / (1.f + nrm2) / (n + 1e-8f);
  if (final_len) {
    vout[idx] = n * sc;   // || squash(s) ||
  } else {
#pragma unroll
    for (int d = 0; d < ODIM; ++d) vout[(size_t)idx * ODIM + d] = tmp[d] * sc;
  }
}

// ---------------------------------------------------------------------------
// Kernel 5: agreement update  L[o,i,b] (+)= <v[b,o,:], x_hat[b,o,i,:]>,
// x_hat recomputed from f32 W (LDS-staged tile) and xsT. Batched 16x8 matvec.
// ---------------------------------------------------------------------------
template <int INIT>
__global__ void k_agree(const float* __restrict__ W,
                        const float* __restrict__ xsT,
                        const float* __restrict__ vin,
                        float* __restrict__ L) {
  __shared__ float Wt[64 * 128];   // 32 KB: W[o, i0..i0+63, 16, 8]
  __shared__ float Vt[128 * 16];   // 8 KB:  v[b, o, :] for all b
  int o     = blockIdx.x >> 5;     // grid = 32 o * 32 itiles
  int itile = blockIdx.x & 31;
  int i0    = itile * 64;
  for (int t = threadIdx.x; t < 64 * 128; t += 256)
    Wt[t] = W[(size_t)(o * ICAPS + i0) * 128 + t];
  for (int t = threadIdx.x; t < 128 * 16; t += 256)
    Vt[t] = vin[((size_t)(t >> 4) * OCAPS + o) * ODIM + (t & 15)];
  __syncthreads();
  for (int task = threadIdx.x; task < 128 * 64; task += 256) {
    int b  = task & 127;
    int il = task >> 7;
    int i  = i0 + il;
    float xv[8];
#pragma unroll
    for (int k = 0; k < 8; ++k) xv[k] = xsT[(size_t)(i * 8 + k) * NB + b];
    const float* wt = Wt + il * 128;
    float dotv = 0.f;
#pragma unroll
    for (int d = 0; d < ODIM; ++d) {
      float xh = 0.f;
#pragma unroll
      for (int k = 0; k < 8; ++k) xh += wt[d * 8 + k] * xv[k];
      dotv += xh * Vt[b * 16 + d];
    }
    size_t li = ((size_t)o * ICAPS + i) * NB + b;
    L[li] = INIT ? dotv : (L[li] + dotv);
  }
}

// ---------------------------------------------------------------------------
// Kernel 6: softmax over the o dimension (32 values) of L[o,i,b] -> cT[o,i,b].
// ---------------------------------------------------------------------------
__global__ void k_softmax(const float* __restrict__ L, float* __restrict__ cT) {
  int idx = blockIdx.x * blockDim.x + threadIdx.x;   // ICAPS*NB = 262144
  if (idx >= ICAPS * NB) return;
  int b = idx & 127;
  int i = idx >> 7;
  float vals[OCAPS];
  float m = -3.4e38f;
#pragma unroll
  for (int o = 0; o < OCAPS; ++o) {
    vals[o] = L[((size_t)o * ICAPS + i) * NB + b];
    m = fmaxf(m, vals[o]);
  }
  float ssum = 0.f;
#pragma unroll
  for (int o = 0; o < OCAPS; ++o) {
    vals[o] = expf(vals[o] - m);
    ssum += vals[o];
  }
  float inv = 1.f / ssum;
#pragma unroll
  for (int o = 0; o < OCAPS; ++o)
    cT[((size_t)o * ICAPS + i) * NB + b] = vals[o] * inv;
}

// ---------------------------------------------------------------------------
extern "C" void kernel_launch(void* const* d_in, const int* in_sizes, int n_in,
                              void* d_out, int out_size, void* d_ws, size_t ws_size,
                              hipStream_t stream) {
  const float* hidden = (const float*)d_in[0];   // [128, 16384]
  const float* cw     = (const float*)d_in[1];   // [8, 8]
  const float* cb     = (const float*)d_in[2];   // [8]
  const float* W      = (const float*)d_in[3];   // [32, 2048, 16, 8]
  float* out = (float*)d_out;                    // [128, 32]
  char* ws = (char*)d_ws;

  // Workspace layout (all L2-resident, ~92.8 MB total):
  float*    xsT = (float*)(ws);                    //  8,388,608 B  x^T [kk][b]
  _Float16* Wh  = (_Float16*)(ws + 8388608);       // 16,777,216 B  packed f16 A
  float*    L   = (float*)(ws + 25165824);         // 33,554,432 B  logits [o][i][b]
  float*    cT  = (float*)(ws + 58720256);         // 33,554,432 B  coeffs [o][i][b]
  float*    s   = (float*)(ws + 92274688);         //    262,144 B  s [b][o][d]
  float*    v   = (float*)(ws + 92536832);         //    262,144 B  v [b][o][d]

  k_primary<<<1024, 256, 0, stream>>>(hidden, cw, cb, xsT);
  k_pack_w<<<2048, 256, 0, stream>>>(W, Wh);

  // routing iteration 0 (uniform c = 1/32, folded into squash scale)
  k_gemm_s<0><<<256, 256, 0, stream>>>(Wh, xsT, cT, s);
  k_squash<<<16, 256, 0, stream>>>(s, v, 1.0f / 32.0f, 0);
  k_agree<1><<<1024, 256, 0, stream>>>(W, xsT, v, L);

  // routing iteration 1
  k_softmax<<<1024, 256, 0, stream>>>(L, cT);
  k_gemm_s<1><<<256, 256, 0, stream>>>(Wh, xsT, cT, s);
  k_squash<<<16, 256, 0, stream>>>(s, v, 1.0f, 0);
  k_agree<0><<<1024, 256, 0, stream>>>(W, xsT, v, L);

  // routing iteration 2 -> final lengths
  k_softmax<<<1024, 256, 0, stream>>>(L, cT);
  k_gemm_s<1><<<256, 256, 0, stream>>>(Wh, xsT, cT, s);
  k_squash<<<16, 256, 0, stream>>>(s, out, 1.0f, 1);
}
